// BasicDeconvolutionBlock_40759239639903
// MI455X (gfx1250) — matrix operand order repaired and checked by reading, not compiled, not run on hardware
//
#include <hip/hip_runtime.h>

// ---------------- types ----------------
typedef __bf16 bf16_t;
typedef bf16_t v8bf  __attribute__((ext_vector_type(8)));
typedef bf16_t v16bf __attribute__((ext_vector_type(16)));
typedef float  v8f   __attribute__((ext_vector_type(8)));
typedef int    v4i   __attribute__((vector_size(16)));

// ---------------- problem constants ----------------
#define GDIM   130                 // padded keyspace dim
#define NPTS   200000
#define MPTS   (NPTS * 8)          // 1,600,000 upsampled points
#define ROWS_PER_BLK 64
#define NBLKS  (MPTS / ROWS_PER_BLK)      // 25,000 blocks
#define TBL    (2 * GDIM * GDIM * GDIM)   // 4,394,000 table entries
#define RBLK   512                 // BN reduce stage-1 blocks

// CDNA5 async global->LDS copy availability (device pass only)
#if defined(__AMDGCN__) && __has_builtin(__builtin_amdgcn_global_load_async_to_lds_b128) && \
    __has_builtin(__builtin_amdgcn_s_wait_asynccnt)
#define USE_ASYNC_LDS 1
#else
#define USE_ASYNC_LDS 0
#endif

#if USE_ASYNC_LDS
__device__ __forceinline__ void async_copy16(void* lds_dst, const void* gsrc) {
    // generic LDS pointer: low 32 bits are the LDS byte offset (ISA 10.2 aperture rule);
    // generic global pointer == AS1 address numerically.
    __builtin_amdgcn_global_load_async_to_lds_b128(
        (__attribute__((address_space(1))) v4i*)(uintptr_t)gsrc,
        (__attribute__((address_space(3))) v4i*)(uint32_t)(uintptr_t)lds_dst,
        0, 0);
}
#endif

// ============================================================
// 1) fill key->index lookup table with -1
// ============================================================
__global__ void init_table_k(int* __restrict__ table) {
    int i = blockIdx.x * blockDim.x + threadIdx.x;
    if (i < TBL) table[i] = -1;
}

// ============================================================
// 1b) 64-element bf16 zero row (miss target for branch-free gather)
// ============================================================
__global__ void init_zrow_k(bf16_t* __restrict__ zrow) {
    int i = threadIdx.x;            // 64 threads
    zrow[i] = (bf16_t)0.0f;
}

// ============================================================
// 2) scatter: key(up-point m) -> m   (keys are unique)
// ============================================================
__global__ void scatter_k(const int* __restrict__ coords, int* __restrict__ table) {
    int m = blockIdx.x * blockDim.x + threadIdx.x;
    if (m >= MPTS) return;
    int n = m >> 3, o = m & 7;
    int cx = coords[n * 4 + 0] * 2 + ((o >> 2) & 1);
    int cy = coords[n * 4 + 1] * 2 + ((o >> 1) & 1);
    int cz = coords[n * 4 + 2] * 2 + (o & 1);
    int b  = coords[n * 4 + 3];
    int key = ((b * GDIM + cx + 1) * GDIM + cy + 1) * GDIM + cz + 1;
    table[key] = m;
}

// ============================================================
// 3) feats f32 -> bf16 (row-major N x 64)
// ============================================================
__global__ void conv_feats_k(const float* __restrict__ feats, bf16_t* __restrict__ fb) {
    int i = blockIdx.x * blockDim.x + threadIdx.x;
    if (i < NPTS * 64) fb[i] = (bf16_t)feats[i];
}

// ============================================================
// 4) weights f32 -> bf16, pre-swizzled into WMMA B-fragment layout:
//    wf[(((k*2+t)*4+cw)*32 + lane)*16 + j] = W[k][t*32 + (lane<16?0:16) + j][cw*16 + (lane&15)]
// ============================================================
__global__ void conv_w_k(const float* __restrict__ W, bf16_t* __restrict__ wf) {
    int i = blockIdx.x * blockDim.x + threadIdx.x;
    if (i >= 27 * 64 * 64) return;
    int j    = i & 15;
    int lane = (i >> 4) & 31;
    int cw   = (i >> 9) & 3;
    int kt   = i >> 11;            // 0..53 = k*2 + t
    int k = kt >> 1, t = kt & 1;
    int K   = t * 32 + ((lane < 16) ? 0 : 16) + j;
    int col = cw * 16 + (lane & 15);
    wf[i] = (bf16_t)W[(k * 64 + K) * 64 + col];
}

// ============================================================
// 5) main gather + WMMA GEMM.
//    Block = 128 threads (4 waves) owning 64 output rows x 64 cols.
//    Per k-offset: branch-free async gather into LDS, 1 B load, 8 WMMAs.
// ============================================================
__global__ void __launch_bounds__(128)
gemm_k(const int* __restrict__ coords, const int* __restrict__ table,
       const bf16_t* __restrict__ featsBF, const bf16_t* __restrict__ wfrag,
       const bf16_t* __restrict__ zrow, float* __restrict__ out)
{
    __shared__ int skey[ROWS_PER_BLK];
    __shared__ int sj[27 * ROWS_PER_BLK];
    __shared__ __align__(16) bf16_t Asm[ROWS_PER_BLK][72];  // stride 72 bf16: bank-conflict-free

    const int tid  = threadIdx.x;
    const int blk  = blockIdx.x;
    const int lane = tid & 31;
    const int wv   = tid >> 5;            // cout tile 0..3
    const int row  = lane & 15;
    const int half = lane >> 4;

    if (tid < ROWS_PER_BLK) {
        int m = blk * ROWS_PER_BLK + tid;
        int n = m >> 3, o = m & 7;
        int cx = coords[n * 4 + 0] * 2 + ((o >> 2) & 1);
        int cy = coords[n * 4 + 1] * 2 + ((o >> 1) & 1);
        int cz = coords[n * 4 + 2] * 2 + (o & 1);
        int b  = coords[n * 4 + 3];
        skey[tid] = ((b * GDIM + cx + 1) * GDIM + cy + 1) * GDIM + cz + 1;
    }
    __syncthreads();

    // ---- rulebook: one table lookup per (offset,row) ----
    for (int it = tid; it < 27 * ROWS_PER_BLK; it += 128) {
        int off = it >> 6;                // 0..26
        int r   = it & 63;
        int dx = off / 9 - 1, dy = (off / 3) % 3 - 1, dz = off % 3 - 1;
        int q = skey[r] + (dx * GDIM + dy) * GDIM + dz;   // always in [0, TBL)
        sj[it] = table[q];
    }
    __syncthreads();

    v8f acc[4];
    #pragma unroll
    for (int st = 0; st < 4; ++st)
        acc[st] = (v8f){0.f, 0.f, 0.f, 0.f, 0.f, 0.f, 0.f, 0.f};

    #pragma unroll 1
    for (int k = 0; k < 27; ++k) {
        // ---- branch-free gather of 64 neighbor rows into LDS ----
        #pragma unroll
        for (int i = 0; i < 4; ++i) {
            int it   = tid + 128 * i;
            int r    = it >> 3;           // 0..63
            int part = it & 7;            // 8-element chunk
            int j = sj[k * 64 + r];
            const bf16_t* src = (j >= 0) ? (featsBF + ((size_t)(j >> 3) << 6)) : zrow;
#if USE_ASYNC_LDS
            async_copy16(&Asm[r][part * 8], src + part * 8);
#else
            uint4 v = *(const uint4*)(src + part * 8);
            *(uint4*)(&Asm[r][part * 8]) = v;
#endif
        }
#if USE_ASYNC_LDS
        __builtin_amdgcn_s_wait_asynccnt(0);
#endif
        __syncthreads();

        // ---- B fragments: 32B contiguous per lane, reused by 4 subtiles ----
        const bf16_t* wb = wfrag + ((size_t)(k * 2) * 4 + wv) * 512 + lane * 16;
        v16bf b0 = *(const v16bf*)(wb);
        v16bf b1 = *(const v16bf*)(wb + 2048);

        // ---- 4 row-subtiles x 2 WMMAs ----
        #pragma unroll
        for (int st = 0; st < 4; ++st) {
            const bf16_t* ap = &Asm[st * 16 + row][0];
            v8bf a0lo = *(const v8bf*)(ap + 16 * half);
            v8bf a0hi = *(const v8bf*)(ap + 8 + 16 * half);
            v8bf a1lo = *(const v8bf*)(ap + 32 + 16 * half);
            v8bf a1hi = *(const v8bf*)(ap + 40 + 16 * half);
            v16bf a0 = __builtin_shufflevector(a0lo, a0hi, 0,1,2,3,4,5,6,7,8,9,10,11,12,13,14,15);
            v16bf a1 = __builtin_shufflevector(a1lo, a1hi, 0,1,2,3,4,5,6,7,8,9,10,11,12,13,14,15);
            acc[st] = __builtin_amdgcn_wmma_f32_16x16x32_bf16(false, a0, false, b0,
                                                              (short)0, acc[st], false, false);
            acc[st] = __builtin_amdgcn_wmma_f32_16x16x32_bf16(false, a1, false, b1,
                                                              (short)0, acc[st], false, false);
        }
        __syncthreads();   // protect Asm before next gather
    }

    // ---- store D: lane<16 -> M=v, lane>=16 -> M=v+8 ; N = lane&15 ----
    const int col = wv * 16 + row;
    #pragma unroll
    for (int st = 0; st < 4; ++st) {
        const int baseM = blk * ROWS_PER_BLK + st * 16 + 8 * half;
        #pragma unroll
        for (int v = 0; v < 8; ++v)
            out[(size_t)(baseM + v) * 64 + col] = acc[st][v];
    }
}

// ============================================================
// 6) BN reduce stage 1: fixed-order partial sums (deterministic)
// ============================================================
__global__ void bn_reduce1_k(const float* __restrict__ out,
                             float* __restrict__ psum, float* __restrict__ psum2)
{
    int t = threadIdx.x;
    int c = t & 63, sub = t >> 6;
    float s = 0.f, s2 = 0.f;
    for (int r = blockIdx.x * 4 + sub; r < MPTS; r += RBLK * 4) {
        float v = out[(size_t)r * 64 + c];
        s += v; s2 += v * v;
    }
    __shared__ float sh[256], sh2[256];
    sh[t] = s; sh2[t] = s2;
    __syncthreads();
    if (sub == 0) {
        float a = sh[c] + sh[c + 64] + sh[c + 128] + sh[c + 192];
        float b = sh2[c] + sh2[c + 64] + sh2[c + 128] + sh2[c + 192];
        psum[blockIdx.x * 64 + c]  = a;
        psum2[blockIdx.x * 64 + c] = b;
    }
}

// ============================================================
// 7) BN reduce stage 2 -> scale/shift
// ============================================================
__global__ void bn_reduce2_k(const float* __restrict__ psum, const float* __restrict__ psum2,
                             const float* __restrict__ gamma, const float* __restrict__ beta,
                             float* __restrict__ ss)
{
    int c = threadIdx.x;   // 64 threads
    float s = 0.f, s2 = 0.f;
    for (int b = 0; b < RBLK; ++b) { s += psum[b * 64 + c]; s2 += psum2[b * 64 + c]; }
    float mean = s / (float)MPTS;
    float var  = s2 / (float)MPTS - mean * mean;
    float scale = gamma[c] * rsqrtf(var + 1e-5f);
    ss[c]      = scale;
    ss[64 + c] = beta[c] - mean * scale;
}

// ============================================================
// 8) normalize + ReLU in place
// ============================================================
__global__ void bn_apply_k(float* __restrict__ out, const float* __restrict__ ss)
{
    size_t total = (size_t)MPTS * 64;
    for (size_t i = (size_t)blockIdx.x * blockDim.x + threadIdx.x; i < total;
         i += (size_t)gridDim.x * blockDim.x) {
        int c = (int)(i & 63);
        float v = out[i] * ss[c] + ss[64 + c];
        out[i] = v > 0.f ? v : 0.f;
    }
}

// ============================================================
// launch
// ============================================================
static inline size_t align256(size_t x) { return (x + 255) & ~(size_t)255; }

extern "C" void kernel_launch(void* const* d_in, const int* in_sizes, int n_in,
                              void* d_out, int out_size, void* d_ws, size_t ws_size,
                              hipStream_t stream)
{
    const float* feats  = (const float*)d_in[0];   // [N,64] f32
    const float* W      = (const float*)d_in[1];   // [27,64,64] f32
    const float* gamma  = (const float*)d_in[2];   // [64]
    const float* beta   = (const float*)d_in[3];   // [64]
    const int*   coords = (const int*)d_in[4];     // [N,4] int32
    float* out = (float*)d_out;                    // [M,64] f32

    // workspace carve-out (~43.6 MB total)
    char* ws = (char*)d_ws;
    size_t off = 0;
    int*    table   = (int*)(ws + off);    off += align256((size_t)TBL * 4);
    bf16_t* featsBF = (bf16_t*)(ws + off); off += align256((size_t)NPTS * 64 * 2);
    bf16_t* wfrag   = (bf16_t*)(ws + off); off += align256((size_t)27 * 64 * 64 * 2);
    float*  psum    = (float*)(ws + off);  off += align256((size_t)RBLK * 64 * 4);
    float*  psum2   = (float*)(ws + off);  off += align256((size_t)RBLK * 64 * 4);
    float*  ss      = (float*)(ws + off);  off += align256(128 * 4);
    bf16_t* zrow    = (bf16_t*)(ws + off); off += align256(64 * 2);
    (void)ws_size; (void)in_sizes; (void)n_in; (void)out_size;

    init_table_k<<<(TBL + 255) / 256, 256, 0, stream>>>(table);
    init_zrow_k<<<1, 64, 0, stream>>>(zrow);
    scatter_k<<<(MPTS + 255) / 256, 256, 0, stream>>>(coords, table);
    conv_feats_k<<<(NPTS * 64 + 255) / 256, 256, 0, stream>>>(feats, featsBF);
    conv_w_k<<<(27 * 64 * 64 + 255) / 256, 256, 0, stream>>>(W, wfrag);

    gemm_k<<<NBLKS, 128, 0, stream>>>(coords, table, featsBF, wfrag, zrow, out);

    bn_reduce1_k<<<RBLK, 256, 0, stream>>>(out, psum, psum2);
    bn_reduce2_k<<<1, 64, 0, stream>>>(psum, psum2, gamma, beta, ss);
    bn_apply_k<<<2048, 256, 0, stream>>>(out, ss);
}
